// InvAttention_11948599018112
// MI455X (gfx1250) — matrix operand order repaired and checked by reading, not compiled, not run on hardware
//
#include <hip/hip_runtime.h>

// Problem constants (match reference)
#define N_    1024
#define E_    8192
#define M_    8192
#define B_    64
#define IN_CH 512
#define H_CH  1024
#define H_    8
#define D_    128

typedef __attribute__((ext_vector_type(16))) __bf16 v16bf;
typedef __attribute__((ext_vector_type(8)))  __bf16 v8bf;
typedef __attribute__((ext_vector_type(8)))  float  v8f;
typedef int v4i __attribute__((vector_size(16)));   // matches async-LDS builtin

#define AS1 __attribute__((address_space(1)))
#define AS3 __attribute__((address_space(3)))

// ---------------------------------------------------------------------------
// CDNA5 async load-to-LDS (ASYNCcnt-tracked), guarded so the file compiles on
// toolchains without the builtins (falls back to plain load/store staging).
// ---------------------------------------------------------------------------
#if __has_builtin(__builtin_amdgcn_global_load_async_to_lds_b128)
#define HAVE_ASYNC_LDS 1
#else
#define HAVE_ASYNC_LDS 0
#endif

#if __has_builtin(__builtin_amdgcn_s_wait_asynccnt)
#define AWAIT_ASYNC(n) __builtin_amdgcn_s_wait_asynccnt(n)
#else
#define AWAIT_ASYNC(n) asm volatile("s_wait_asynccnt %0" ::"i"(n) : "memory")
#endif

// ---------------------------------------------------------------------------
// WMMA helpers (CDNA5 16x16x32 bf16, f32 accumulate)
// ---------------------------------------------------------------------------
__device__ __forceinline__ v8f wmma_bf16(v16bf a, v16bf b, v8f c) {
  return __builtin_amdgcn_wmma_f32_16x16x32_bf16(
      /*neg_a=*/false, a, /*neg_b=*/false, b,
      /*c_mod=*/(short)0, c, /*reuse_a=*/false, /*reuse_b=*/false);
}

// A fragment (16 rows x 32 K, bf16).
//   lane: row = lane&15, kh = lane>>4
//   element i -> K = kh*16 + i
__device__ __forceinline__ v16bf frag_A_bf16(const __bf16* row, int kh) {
  const __bf16* p = row + kh * 16;
  v8bf lo = *(const v8bf*)(p);
  v8bf hi = *(const v8bf*)(p + 8);
  v16bf r;
#pragma unroll
  for (int i = 0; i < 8; ++i) { r[i] = lo[i]; r[8 + i] = hi[i]; }
  return r;
}

__device__ __forceinline__ v16bf frag_A_f32(const float* row, int kh) {
  const float* p = row + kh * 16;
  v16bf r;
#pragma unroll
  for (int i = 0; i < 8; ++i) { r[i] = (__bf16)p[i]; r[8 + i] = (__bf16)p[8 + i]; }
  return r;
}

// B fragment (32 K x 16 cols): lane: col = lane&15, kh = lane>>4,
// element i -> K = kh*16 + i.  colp = K-contiguous source row for this column.
__device__ __forceinline__ v16bf frag_B_bf16(const __bf16* colp, int kh) {
  const __bf16* p = colp + kh * 16;
  v8bf lo = *(const v8bf*)(p);
  v8bf hi = *(const v8bf*)(p + 8);
  v16bf r;
#pragma unroll
  for (int i = 0; i < 8; ++i) { r[i] = lo[i]; r[8 + i] = hi[i]; }
  return r;
}

__device__ __forceinline__ v16bf frag_B_f32(const float* colp, int kh) {
  const float* p = colp + kh * 16;
  v16bf r;
#pragma unroll
  for (int i = 0; i < 16; ++i) r[i] = (__bf16)p[i];
  return r;
}

// Order-preserving float<->uint encoding for atomic max of signed floats.
__device__ __forceinline__ unsigned enc_f(float x) {
  unsigned u = __float_as_uint(x);
  return (u & 0x80000000u) ? ~u : (u | 0x80000000u);
}
__device__ __forceinline__ float dec_f(unsigned u) {
  return (u & 0x80000000u) ? __uint_as_float(u & 0x7fffffffu)
                           : __uint_as_float(~u);
}

// ---------------------------------------------------------------------------
// Score tile: S[e0:e0+16, n0:n0+16] for one head.
//   S = Ke (16xD, rows gathered via atom_index) x Qh^T (Dx16), q pre-scaled.
// C/D layout: element (row=v+8*kh, col=lane&15).
// ---------------------------------------------------------------------------
__device__ __forceinline__ v8f score_tile(const __bf16* __restrict__ qh_head,
                                          const __bf16* __restrict__ kh_head,
                                          const int* __restrict__ atom_index,
                                          int e0, int n0, int lane) {
  const int r = lane & 15, kh = lane >> 4;
  const __bf16* ap = kh_head + (size_t)atom_index[e0 + r] * D_;   // A row = edge
  const __bf16* bp = qh_head + (size_t)(n0 + r) * D_;             // B col = node
  v8f c = {};
#pragma unroll
  for (int k0 = 0; k0 < D_; k0 += 32)
    c = wmma_bf16(frag_A_bf16(ap + k0, kh), frag_B_bf16(bp + k0, kh), c);
  return c;
}

// ---------------------------------------------------------------------------
// K1: fused q/k/v projections.  out = (X @ W^T + b) [*1/sqrt(D) for q],
// stored bf16 head-major [H][N][D].
// ---------------------------------------------------------------------------
__global__ void k_proj(const float* __restrict__ q, const float* __restrict__ k,
                       const float* __restrict__ v,
                       const float* __restrict__ Wq, const float* __restrict__ bq,
                       const float* __restrict__ Wk, const float* __restrict__ bk,
                       const float* __restrict__ Wv, const float* __restrict__ bv,
                       __bf16* __restrict__ qhb, __bf16* __restrict__ khb,
                       __bf16* __restrict__ vhb) {
  const int which = blockIdx.z;
  const float* X  = (which == 0) ? q  : (which == 1) ? k  : v;
  const float* W  = (which == 0) ? Wq : (which == 1) ? Wk : Wv;
  const float* bi = (which == 0) ? bq : (which == 1) ? bk : bv;
  __bf16* dst     = (which == 0) ? qhb : (which == 1) ? khb : vhb;
  const float scale = (which == 0) ? 0.08838834764831843f : 1.0f;  // 1/sqrt(128)

  const int lane = threadIdx.x;
  const int r = lane & 15, kh = lane >> 4;
  const int r0 = blockIdx.x * 16;   // output row tile (node n)
  const int c0 = blockIdx.y * 16;   // output col tile (channel o)

  v8f c = {};
  const float* arow = X + (size_t)(r0 + r) * IN_CH;
  const float* bcol = W + (size_t)(c0 + r) * IN_CH;   // B col o == W row o
#pragma unroll
  for (int k0 = 0; k0 < IN_CH; k0 += 32)
    c = wmma_bf16(frag_A_f32(arow + k0, kh), frag_B_f32(bcol + k0, kh), c);

  const int col = c0 + r;
  const float bcolv = bi[col];
  const int hh = col >> 7, d = col & 127;
#pragma unroll
  for (int vv = 0; vv < 8; ++vv) {
    const int n = r0 + vv + 8 * kh;
    dst[((size_t)hh * N_ + n) * D_ + d] = (__bf16)((c[vv] + bcolv) * scale);
  }
}

// ---------------------------------------------------------------------------
// K-init: seg_max(enc) = enc(-inf), seg_sum = 0
// ---------------------------------------------------------------------------
__global__ void k_init(unsigned* __restrict__ segmax, float* __restrict__ segsum) {
  const int i = blockIdx.x * blockDim.x + threadIdx.x;
  if (i < B_ * H_ * N_) { segmax[i] = 0x007FFFFFu; segsum[i] = 0.0f; }
}

// ---------------------------------------------------------------------------
// K2a: segment max of (score + bias) into seg_max[B][H][N].
// batch_index is sorted; a lane's 8 score rows are 8 consecutive edges, so
// pre-reduce in registers and emit ONE atomic when they share a segment.
// ---------------------------------------------------------------------------
__global__ void k_seg_max(const __bf16* __restrict__ qhb, const __bf16* __restrict__ khb,
                          const int* __restrict__ atom_index,
                          const int* __restrict__ batch_index,
                          const int* __restrict__ edge_map,
                          const float* __restrict__ attn_bias,
                          unsigned* __restrict__ segmax) {
  const int lane = threadIdx.x;
  const int r = lane & 15, kh = lane >> 4;
  const int e0 = blockIdx.x * 16, n0 = blockIdx.y * 16, h = blockIdx.z;
  __builtin_prefetch(atom_index + e0 + 16, 0, 1);
  v8f s = score_tile(qhb + (size_t)h * N_ * D_, khb + (size_t)h * N_ * D_,
                     atom_index, e0, n0, lane);
  const int n = n0 + r;
  float sv[8];
#pragma unroll
  for (int vv = 0; vv < 8; ++vv)
    sv[vv] = s[vv] + attn_bias[(size_t)n * M_ + edge_map[e0 + vv + 8 * kh]];

  const int b0 = batch_index[e0 + 8 * kh];
  const int b7 = batch_index[e0 + 8 * kh + 7];
  if (b0 == b7) {                 // common case: one segment for this lane
    float m = sv[0];
#pragma unroll
    for (int vv = 1; vv < 8; ++vv) m = fmaxf(m, sv[vv]);
    atomicMax(&segmax[((size_t)b0 * H_ + h) * N_ + n], enc_f(m));
  } else {                        // segment boundary inside the 8 edges
#pragma unroll
    for (int vv = 0; vv < 8; ++vv) {
      const int b = batch_index[e0 + vv + 8 * kh];
      atomicMax(&segmax[((size_t)b * H_ + h) * N_ + n], enc_f(sv[vv]));
    }
  }
}

// ---------------------------------------------------------------------------
// K2b: seg_sum += exp(s - seg_max) * env, with the same lane pre-reduction.
// ---------------------------------------------------------------------------
__global__ void k_seg_sum(const __bf16* __restrict__ qhb, const __bf16* __restrict__ khb,
                          const int* __restrict__ atom_index,
                          const int* __restrict__ batch_index,
                          const int* __restrict__ edge_map,
                          const float* __restrict__ attn_bias,
                          const float* __restrict__ envelope,
                          const unsigned* __restrict__ segmax,
                          float* __restrict__ segsum) {
  const int lane = threadIdx.x;
  const int r = lane & 15, kh = lane >> 4;
  const int e0 = blockIdx.x * 16, n0 = blockIdx.y * 16, h = blockIdx.z;
  __builtin_prefetch(atom_index + e0 + 16, 0, 1);
  v8f s = score_tile(qhb + (size_t)h * N_ * D_, khb + (size_t)h * N_ * D_,
                     atom_index, e0, n0, lane);
  const int n = n0 + r;
  float sv[8], ev[8];
#pragma unroll
  for (int vv = 0; vv < 8; ++vv) {
    const int em = edge_map[e0 + vv + 8 * kh];
    sv[vv] = s[vv] + attn_bias[(size_t)n * M_ + em];
    ev[vv] = envelope[em];
  }
  const int b0 = batch_index[e0 + 8 * kh];
  const int b7 = batch_index[e0 + 8 * kh + 7];
  if (b0 == b7) {
    const size_t si = ((size_t)b0 * H_ + h) * N_ + n;
    const float m = dec_f(segmax[si]);
    float acc = 0.0f;
#pragma unroll
    for (int vv = 0; vv < 8; ++vv) acc += __expf(sv[vv] - m) * ev[vv];
    atomicAdd(&segsum[si], acc);
  } else {
#pragma unroll
    for (int vv = 0; vv < 8; ++vv) {
      const int b = batch_index[e0 + vv + 8 * kh];
      const size_t si = ((size_t)b * H_ + h) * N_ + n;
      atomicAdd(&segsum[si], __expf(sv[vv] - dec_f(segmax[si])) * ev[vv]);
    }
  }
}

// ---------------------------------------------------------------------------
// K3: flash-style fused pass.  Per (head, 16-node tile): loop edge chunks of
// 32; the gathered 32x128 V tile is staged into LDS (async load-to-LDS with
// double buffering when available) while the score tile + softmax weights are
// computed; then 8 WMMAs accumulate the [16n x 128d] output tile in registers.
// ---------------------------------------------------------------------------
#define VROW (D_ + 8)   // pad LDS rows (16B) to spread banks

__device__ __forceinline__ void stage_v_tile(const __bf16* __restrict__ vh_head,
                                             const int* __restrict__ atom_index,
                                             int e0, __bf16 (*buf)[VROW], int lane) {
  const __bf16* vrow = vh_head + (size_t)atom_index[e0 + lane] * D_;
#if HAVE_ASYNC_LDS
#pragma unroll
  for (int j = 0; j < 16; ++j)
    __builtin_amdgcn_global_load_async_to_lds_b128(
        (AS1 v4i*)(vrow + j * 8),
        (AS3 v4i*)&buf[lane][j * 8],
        /*offset=*/0, /*cpol=*/0);
#else
  const uint4* gs = (const uint4*)vrow;
#pragma unroll
  for (int j = 0; j < 16; ++j) *(uint4*)&buf[lane][j * 8] = gs[j];
#endif
}

__global__ void k_attn_out(const __bf16* __restrict__ qhb, const __bf16* __restrict__ khb,
                           const __bf16* __restrict__ vhb,
                           const int* __restrict__ atom_index,
                           const int* __restrict__ batch_index,
                           const int* __restrict__ edge_map,
                           const float* __restrict__ attn_bias,
                           const float* __restrict__ envelope,
                           const unsigned* __restrict__ segmax,
                           const float* __restrict__ segsum,
                           float* __restrict__ attn_out) {
  __shared__ __align__(64) __bf16 wlds[16][32];       // w^T tile [n][e]
  __shared__ __align__(64) __bf16 vlds[2][32][VROW];  // double-buffered V tile
  const int lane = threadIdx.x;
  const int r = lane & 15, kh = lane >> 4;
  const int n0 = blockIdx.x * 16, h = blockIdx.y;
  const __bf16* qh_head = qhb + (size_t)h * N_ * D_;
  const __bf16* kh_head = khb + (size_t)h * N_ * D_;
  const __bf16* vh_head = vhb + (size_t)h * N_ * D_;
  const int n = n0 + r;

  v8f acc[8] = {};

  stage_v_tile(vh_head, atom_index, 0, vlds[0], lane);   // prime pipeline

  for (int e0 = 0; e0 < E_; e0 += 32) {
    const int cur = (e0 >> 5) & 1;
    const bool more = (e0 + 32) < E_;
    if (more) {                               // overlap next copy with compute
      __builtin_prefetch(atom_index + e0 + 64, 0, 1);
      stage_v_tile(vh_head, atom_index, e0 + 32, vlds[cur ^ 1], lane);
    }

#pragma unroll
    for (int half = 0; half < 2; ++half) {
      const int eb = e0 + half * 16;
      v8f s = score_tile(qh_head, kh_head, atom_index, eb, n0, lane);
#pragma unroll
      for (int vv = 0; vv < 8; ++vv) {
        const int e  = eb + vv + 8 * kh;
        const int em = edge_map[e];
        const int b  = batch_index[e];
        const float env = envelope[em];
        const float svv = s[vv] + attn_bias[(size_t)n * M_ + em];
        const size_t si = ((size_t)b * H_ + h) * N_ + n;
        const float w = env * env * __expf(svv - dec_f(segmax[si])) /
                        (segsum[si] + 1e-16f);
        wlds[r][half * 16 + vv + 8 * kh] = (__bf16)w;   // transpose to [n][e]
      }
    }

#if HAVE_ASYNC_LDS
    if (more) AWAIT_ASYNC(16);   // current tile's 16 loads retired (in-order)
    else      AWAIT_ASYNC(0);
#endif
    __syncthreads();

    // A = w^T (16 n rows x 32 e K) straight out of LDS
    const v16bf aw = frag_A_bf16(&wlds[r][0], kh);
#pragma unroll
    for (int dt = 0; dt < 8; ++dt) {
      v16bf bv;   // B[e][d]: col d = dt*16 + (lane&15), K = edge in chunk
#pragma unroll
      for (int i = 0; i < 16; ++i)
        bv[i] = vlds[cur][kh * 16 + i][dt * 16 + r];
      acc[dt] = wmma_bf16(aw, bv, acc[dt]);
    }
    __syncthreads();
  }

#pragma unroll
  for (int dt = 0; dt < 8; ++dt)
#pragma unroll
    for (int vv = 0; vv < 8; ++vv)
      attn_out[(size_t)(n0 + vv + 8 * kh) * H_CH + h * D_ + dt * 16 + r] =
          acc[dt][vv];
}

// ---------------------------------------------------------------------------
// K4a: LayerNorm over H_CH (eps = 1e-7, biased var), write bf16 for final GEMM
// ---------------------------------------------------------------------------
__global__ void k_layernorm(const float* __restrict__ x, const float* __restrict__ g,
                            const float* __restrict__ bta, __bf16* __restrict__ y) {
  const int n = blockIdx.x, t = threadIdx.x;          // 256 threads
  __shared__ float rs[256], rs2[256];
  float xv[4], s = 0.f, s2 = 0.f;
#pragma unroll
  for (int j = 0; j < 4; ++j) {
    xv[j] = x[(size_t)n * H_CH + t + j * 256];
    s += xv[j]; s2 += xv[j] * xv[j];
  }
  rs[t] = s; rs2[t] = s2; __syncthreads();
  for (int o = 128; o > 0; o >>= 1) {
    if (t < o) { rs[t] += rs[t + o]; rs2[t] += rs2[t + o]; }
    __syncthreads();
  }
  const float mean = rs[0] * (1.0f / H_CH);
  const float var  = rs2[0] * (1.0f / H_CH) - mean * mean;
  const float rstd = rsqrtf(var + 1e-7f);
#pragma unroll
  for (int j = 0; j < 4; ++j) {
    const int c = t + j * 256;
    y[(size_t)n * H_CH + c] = (__bf16)((xv[j] - mean) * rstd * g[c] + bta[c]);
  }
}

// ---------------------------------------------------------------------------
// K4b: out = ynorm @ Wo^T + bo  -> d_out [N x IN_CH] fp32
// ---------------------------------------------------------------------------
__global__ void k_out_proj(const __bf16* __restrict__ ynorm,
                           const float* __restrict__ Wo,
                           const float* __restrict__ bo,
                           float* __restrict__ out) {
  const int lane = threadIdx.x;
  const int r = lane & 15, kh = lane >> 4;
  const int n0 = blockIdx.x * 16, o0 = blockIdx.y * 16;
  v8f c = {};
  const __bf16* arow = ynorm + (size_t)(n0 + r) * H_CH;
  const float*  bcol = Wo + (size_t)(o0 + r) * H_CH;
#pragma unroll
  for (int k0 = 0; k0 < H_CH; k0 += 32)
    c = wmma_bf16(frag_A_bf16(arow + k0, kh), frag_B_f32(bcol + k0, kh), c);
  const float bias = bo[o0 + r];
#pragma unroll
  for (int vv = 0; vv < 8; ++vv)
    out[(size_t)(n0 + vv + 8 * kh) * IN_CH + o0 + r] = c[vv] + bias;
}

// ---------------------------------------------------------------------------
// Host launch
// ---------------------------------------------------------------------------
extern "C" void kernel_launch(void* const* d_in, const int* in_sizes, int n_in,
                              void* d_out, int out_size, void* d_ws, size_t ws_size,
                              hipStream_t stream) {
  (void)in_sizes; (void)n_in; (void)out_size; (void)ws_size;
  const float* q         = (const float*)d_in[0];
  const float* k         = (const float*)d_in[1];
  const float* v         = (const float*)d_in[2];
  const float* envelope  = (const float*)d_in[3];
  const float* attn_bias = (const float*)d_in[4];
  const float* Wq        = (const float*)d_in[5];
  const float* bq        = (const float*)d_in[6];
  const float* Wk        = (const float*)d_in[7];
  const float* bk        = (const float*)d_in[8];
  const float* Wv        = (const float*)d_in[9];
  const float* bv        = (const float*)d_in[10];
  const float* ln_g      = (const float*)d_in[11];
  const float* ln_b      = (const float*)d_in[12];
  const float* Wo        = (const float*)d_in[13];
  const float* bo        = (const float*)d_in[14];
  const int* atom_index  = (const int*)d_in[15];
  const int* batch_index = (const int*)d_in[16];
  const int* edge_map    = (const int*)d_in[17];

  char* ws = (char*)d_ws;
  __bf16*   qhb      = (__bf16*)(ws + (0u  << 20));   // [H][N][D] bf16, 2 MB
  __bf16*   khb      = (__bf16*)(ws + (2u  << 20));   // 2 MB
  __bf16*   vhb      = (__bf16*)(ws + (4u  << 20));   // 2 MB
  __bf16*   ynorm    = (__bf16*)(ws + (6u  << 20));   // [N][H_CH] bf16, 2 MB
  unsigned* segmax   = (unsigned*)(ws + (8u  << 20)); // [B][H][N] enc, 2 MB
  float*    segsum   = (float*)(ws + (10u << 20));    // [B][H][N], 2 MB
  float*    attn_out = (float*)(ws + (12u << 20));    // [N][H_CH] f32, 4 MB

  k_proj<<<dim3(N_ / 16, H_CH / 16, 3), 32, 0, stream>>>(
      q, k, v, Wq, bq, Wk, bk, Wv, bv, qhb, khb, vhb);
  k_init<<<(B_ * H_ * N_) / 256, 256, 0, stream>>>(segmax, segsum);
  k_seg_max<<<dim3(E_ / 16, N_ / 16, H_), 32, 0, stream>>>(
      qhb, khb, atom_index, batch_index, edge_map, attn_bias, segmax);
  k_seg_sum<<<dim3(E_ / 16, N_ / 16, H_), 32, 0, stream>>>(
      qhb, khb, atom_index, batch_index, edge_map, attn_bias, envelope,
      segmax, segsum);
  k_attn_out<<<dim3(N_ / 16, H_), 32, 0, stream>>>(
      qhb, khb, vhb, atom_index, batch_index, edge_map, attn_bias, envelope,
      segmax, segsum, attn_out);
  k_layernorm<<<N_, 256, 0, stream>>>(attn_out, ln_g, ln_b, ynorm);
  k_out_proj<<<dim3(N_ / 16, IN_CH / 16), 32, 0, stream>>>(
      ynorm, Wo, bo, (float*)d_out);
}
